// SparseCnnUp1_34265249087499
// MI455X (gfx1250) — matrix operand order repaired and checked by reading, not compiled, not run on hardware
//
#include <hip/hip_runtime.h>

typedef __attribute__((ext_vector_type(16))) _Float16 v16h;
typedef __attribute__((ext_vector_type(8)))  _Float16 v8h;
typedef __attribute__((ext_vector_type(8)))  float    v8f;

#define N_IN_V   40000
#define N_UP_V   320000
#define N_KEEP_V 192000

// ---------------------------------------------------------------------------
// Pack f32 weights [nK][Cin][Cout] into per-lane B-fragment layout (f16).
// Fragment = 32x16 (KxN), frag index = k*KT*NT + kt*NT + nt.
// B layout (16-bit, 32x16): N = lane%16; element h of the lane's v16h is
// K = 8*(lane>=16)+h for h<8, K = 16+8*(lane>=16)+(h-8) for h>=8.
// K >= Cin (padded 16-channel tensors) packs zeros.
// ---------------------------------------------------------------------------
__global__ void pack_b_frags(const float* __restrict__ src, _Float16* __restrict__ dst,
                             int Cin, int Cout, int KT, int NT) {
    int frag = blockIdx.x;
    int lane = threadIdx.x;          // one wave per fragment
    int nt =  frag % NT;
    int kt = (frag / NT) % KT;
    int k  =  frag / (NT * KT);
    _Float16* out = dst + ((size_t)frag * 32 + lane) * 16;
    int n = nt * 16 + (lane & 15);
    int kbase = kt * 32 + ((lane >> 4) ? 8 : 0);
#pragma unroll
    for (int h = 0; h < 16; ++h) {
        int K = kbase + h + (h >= 8 ? 8 : 0);
        float v = (K < Cin) ? src[((size_t)k * Cin + K) * Cout + n] : 0.0f;
        out[h] = (_Float16)v;
    }
}

__global__ void cvt_f32_to_f16(const float* __restrict__ src, _Float16* __restrict__ dst, int n) {
    int i = blockIdx.x * blockDim.x + threadIdx.x;
    if (i < n) dst[i] = (_Float16)src[i];
}

__global__ void zero_f16_k(_Float16* __restrict__ dst, int n) {
    int i = blockIdx.x * blockDim.x + threadIdx.x;
    if (i < n) dst[i] = (_Float16)0.0f;
}

// ---------------------------------------------------------------------------
// WMMA gather-GEMM: one wave -> a 32-row x (NT*16)-col output tile.
// Each gathered A-fragment pair is reused across all NT column tiles; the
// 2*NT independent accumulator chains keep the matrix pipe saturated.
//   out[row,:] = act( sum_k  xin[nidx[row,k],:] @ W[k]  + bias  (+ residual) )
// GATHER=false -> dense identity row map (single k offset).
// Output row map: outRow = inRow*rowScale + rowOffset (tconv child scatter).
// ---------------------------------------------------------------------------
template<int NT, int KT, bool GATHER>
__global__ __launch_bounds__(256) void sconv_wmma(
    const _Float16* __restrict__ xin, int strideIn,
    const int* __restrict__ nidx,
    const _Float16* __restrict__ bfrag,
    const float* __restrict__ bias,
    const _Float16* __restrict__ residual,          // stride 64, may be null
    _Float16* __restrict__ out16, int strideOut,
    int colOffset, int doRelu, int padHi,
    int rowTiles, int rowScale, int rowOffset)      // rowTiles in units of 32 rows
{
    constexpr int NOFF = GATHER ? 27 : 1;
    int wave = (int)((blockIdx.x * blockDim.x + threadIdx.x) >> 5);
    int lane = threadIdx.x & 31;
    if (wave >= rowTiles) return;
    int m  = lane & 15;
    int hi = lane >> 4;
    int hoff = hi * 8;
    int rowA0 = wave * 32 + m;
    int rowA1 = rowA0 + 16;

    v8f acc[NT][2];
#pragma unroll
    for (int nt = 0; nt < NT; ++nt) { acc[nt][0] = {}; acc[nt][1] = {}; }

    const _Float16* bf = bfrag + (size_t)lane * 16;
    for (int k = 0; k < NOFF; ++k) {
        int srcRow0 = GATHER ? nidx[(size_t)rowA0 * NOFF + k] : rowA0;
        int srcRow1 = GATHER ? nidx[(size_t)rowA1 * NOFF + k] : rowA1;
        const _Float16* xr0 = xin + (size_t)srcRow0 * strideIn + hoff;
        const _Float16* xr1 = xin + (size_t)srcRow1 * strideIn + hoff;
#pragma unroll
        for (int kt = 0; kt < KT; ++kt) {
            union { v16h v; v8h h[2]; } a0, a1;
            a0.h[0] = *(const v8h*)(xr0 + kt * 32);
            a0.h[1] = *(const v8h*)(xr0 + kt * 32 + 16);
            a1.h[0] = *(const v8h*)(xr1 + kt * 32);
            a1.h[1] = *(const v8h*)(xr1 + kt * 32 + 16);
#pragma unroll
            for (int nt = 0; nt < NT; ++nt) {
                v16h b = *(const v16h*)(bf + (size_t)(kt * NT + nt) * 512);
                acc[nt][0] = __builtin_amdgcn_wmma_f32_16x16x32_f16(
                    false, a0.v, false, b, (short)0, acc[nt][0], false, false);
                acc[nt][1] = __builtin_amdgcn_wmma_f32_16x16x32_f16(
                    false, a1.v, false, b, (short)0, acc[nt][1], false, false);
            }
        }
        bf += (size_t)KT * NT * 512;
    }

    int n = lane & 15;
#pragma unroll
    for (int nt = 0; nt < NT; ++nt) {
        int col = colOffset + nt * 16 + n;
        float bv = bias[nt * 16 + n];
#pragma unroll
        for (int half = 0; half < 2; ++half) {
#pragma unroll
            for (int r = 0; r < 8; ++r) {
                int row = (wave * 32 + half * 16 + hi * 8 + r) * rowScale + rowOffset;
                float v = acc[nt][half][r] + bv;
                if (residual) v += (float)residual[(size_t)row * 64 + col];
                if (doRelu) v = fmaxf(v, 0.0f);
                out16[(size_t)row * strideOut + col] = (_Float16)v;
                if (padHi) out16[(size_t)row * strideOut + col + 16] = (_Float16)0.0f;
            }
        }
    }
}

// Final prune: out[i,:] = x[keep_idx[i],:] as f32
__global__ void gather_keep(const _Float16* __restrict__ x, const int* __restrict__ keep,
                            float* __restrict__ out, int nKeep) {
    int i = blockIdx.x * blockDim.x + threadIdx.x;
    int row = i >> 6, c = i & 63;
    if (row < nKeep) out[i] = (float)x[(size_t)keep[row] * 64 + c];
}

extern "C" void kernel_launch(void* const* d_in, const int* in_sizes, int n_in,
                              void* d_out, int out_size, void* d_ws, size_t ws_size,
                              hipStream_t stream) {
    const float* feats = (const float*)d_in[0];
    const float* wt = (const float*)d_in[1];   const float* bt = (const float*)d_in[2];
    const float* w0 = (const float*)d_in[3];   const float* b0 = (const float*)d_in[4];
    const float* wa = (const float*)d_in[5];   const float* ba = (const float*)d_in[6];
    const float* wb = (const float*)d_in[7];   const float* bb = (const float*)d_in[8];
    const float* wc = (const float*)d_in[9];   const float* bc = (const float*)d_in[10];
    const float* wd = (const float*)d_in[11];  const float* bd = (const float*)d_in[12];
    const float* we = (const float*)d_in[13];  const float* be = (const float*)d_in[14];
    const int* n_idx   = (const int*)d_in[15];
    const int* keep    = (const int*)d_in[16];
    float* out = (float*)d_out;

    // ---- workspace carve (all f16 activations; everything fits in 192MB L2) --
    char* p = (char*)d_ws;
    auto carve = [&](size_t bytes) { char* r = p; p += (bytes + 255) & ~(size_t)255; return r; };
    _Float16* h64a   = (_Float16*)carve((size_t)(N_UP_V + 1) * 64 * 2);
    _Float16* h64b   = (_Float16*)carve((size_t)(N_UP_V + 1) * 64 * 2);
    _Float16* h32a   = (_Float16*)carve((size_t)(N_UP_V + 1) * 32 * 2);
    _Float16* h32b   = (_Float16*)carve((size_t)(N_UP_V + 1) * 32 * 2);
    _Float16* featsh = (_Float16*)carve((size_t)N_IN_V * 32 * 2);
    _Float16* wp     = (_Float16*)carve((size_t)665 * 512 * 2);

    // fragment bases (units of 512-half fragments)
    _Float16* wt_p = wp;                      // 8*1*4  = 32 frags
    _Float16* w0_p = wp + (size_t)32 * 512;   // 27*2*4 = 216
    auto wa_p = [&](int i){ return wp + (size_t)(248 + i * 54) * 512; }; // 27*2*1
    auto wb_p = [&](int i){ return wp + (size_t)(410 + i * 54) * 512; }; // 27*1*2
    auto wc_p = [&](int i){ return wp + (size_t)(572 + i * 2 ) * 512; }; // 1*2*1
    auto wd_p = [&](int i){ return wp + (size_t)(578 + i * 27) * 512; }; // 27*1*1
    auto we_p = [&](int i){ return wp + (size_t)(659 + i * 2 ) * 512; }; // 1*1*2

    // ---- pack weights to WMMA B layout (f16) -------------------------------
    pack_b_frags<<<32, 32, 0, stream>>>(wt, wt_p, 32, 64, 1, 4);
    pack_b_frags<<<216, 32, 0, stream>>>(w0, w0_p, 64, 64, 2, 4);
    for (int i = 0; i < 3; ++i) {
        pack_b_frags<<<54, 32, 0, stream>>>(wa + (size_t)i*27*64*16, wa_p(i), 64, 16, 2, 1);
        pack_b_frags<<<54, 32, 0, stream>>>(wb + (size_t)i*27*16*32, wb_p(i), 16, 32, 1, 2);
        pack_b_frags<<< 2, 32, 0, stream>>>(wc + (size_t)i*64*16,    wc_p(i), 64, 16, 2, 1);
        pack_b_frags<<<27, 32, 0, stream>>>(wd + (size_t)i*27*16*16, wd_p(i), 16, 16, 1, 1);
        pack_b_frags<<< 2, 32, 0, stream>>>(we + (size_t)i*16*32,    we_p(i), 16, 32, 1, 2);
    }

    // feats -> f16; zero the "no neighbor" pad rows
    cvt_f32_to_f16<<<(N_IN_V * 32) / 256, 256, 0, stream>>>(feats, featsh, N_IN_V * 32);
    zero_f16_k<<<1, 64, 0, stream>>>(h64a + (size_t)N_UP_V * 64, 64);
    zero_f16_k<<<1, 64, 0, stream>>>(h64b + (size_t)N_UP_V * 64, 64);
    zero_f16_k<<<1, 32, 0, stream>>>(h32a + (size_t)N_UP_V * 32, 32);
    zero_f16_k<<<1, 32, 0, stream>>>(h32b + (size_t)N_UP_V * 32, 32);

    // one wave per 32-row tile
    auto blocks = [](int rowTiles32) { return (rowTiles32 * 32 + 255) / 256; };
    const int PT = N_IN_V / 32;   // 1250 parent tiles
    const int RT = N_UP_V / 32;   // 10000 row tiles

    // ---- transpose conv k=2,s=2: child p*8+k = relu(feats[p] @ wt[k] + bt) --
    for (int k = 0; k < 8; ++k) {
        sconv_wmma<4, 1, false><<<blocks(PT), 256, 0, stream>>>(
            featsh, 32, nullptr, wt_p + (size_t)k * 4 * 512,
            bt, nullptr, h64a, 64, 0, 1, 0, PT, 8, k);
    }

    // ---- conv0: x = relu(sconv3(x, w0, b0)) --------------------------------
    sconv_wmma<4, 2, true><<<blocks(RT), 256, 0, stream>>>(
        h64a, 64, n_idx, w0_p, b0, nullptr, h64b, 64, 0, 1, 0, RT, 1, 0);

    // ---- 3 InceptionResNet blocks ------------------------------------------
    _Float16* cur = h64b;
    for (int i = 0; i < 3; ++i) {
        _Float16* nxt = (cur == h64b) ? h64a : h64b;
        // p0 path: t1 = relu(sconv(x, wa)) : 64 -> 16 (stride-32, pad hi)
        sconv_wmma<1, 2, true><<<blocks(RT), 256, 0, stream>>>(
            cur, 64, n_idx, wa_p(i), ba + i*16, nullptr,
            h32a, 32, 0, 1, 1, RT, 1, 0);
        // p0 = sconv(t1, wb) + bb + x[:,0:32]   (writes nxt cols 0..31)
        sconv_wmma<2, 1, true><<<blocks(RT), 256, 0, stream>>>(
            h32a, 32, n_idx, wb_p(i), bb + i*32, cur,
            nxt, 64, 0, 0, 0, RT, 1, 0);
        // p1 path: t2 = relu(x @ wc + bc) : dense 64 -> 16
        sconv_wmma<1, 2, false><<<blocks(RT), 256, 0, stream>>>(
            cur, 64, nullptr, wc_p(i), bc + i*16, nullptr,
            h32b, 32, 0, 1, 1, RT, 1, 0);
        // t3 = relu(sconv(t2, wd)) : 16 -> 16
        sconv_wmma<1, 1, true><<<blocks(RT), 256, 0, stream>>>(
            h32b, 32, n_idx, wd_p(i), bd + i*16, nullptr,
            h32a, 32, 0, 1, 1, RT, 1, 0);
        // p1 = t3 @ we + be + x[:,32:64]   (writes nxt cols 32..63)
        sconv_wmma<2, 1, false><<<blocks(RT), 256, 0, stream>>>(
            h32a, 32, nullptr, we_p(i), be + i*32, cur,
            nxt, 64, 32, 0, 0, RT, 1, 0);
        cur = nxt;
    }

    // ---- prune: gather keep_idx rows to f32 output -------------------------
    gather_keep<<<(N_KEEP_V * 64) / 256, 256, 0, stream>>>(cur, keep, out, N_KEEP_V);
}
